// RoteLearningMemory_80152679678780
// MI455X (gfx1250) — compile-verified, matrix operand order checked by reading, not run on hardware
//
#include <hip/hip_runtime.h>
#include <hip/hip_bf16.h>
#include <cstdint>

// ---------------------------------------------------------------------------
// RoteLearningMemory forward for MI455X (gfx1250, wave32, WMMA).
// Big GEMMs: 128x128 block tile, 8 waves, wave tile 32x64 -> 8
// v_wmma_f32_16x16x32_bf16 per K-step, software-pipelined global->LDS
// staging (fp32 -> bf16 via v_cvt_pk_bf16_f32) + global_prefetch_b8.
// ---------------------------------------------------------------------------

#define B_      32
#define S_      2048
#define D_IN_   1024
#define M_SIZE_ 256
#define M_CELLS_ 16
#define N_HEADS_ 4
#define HEAD_DIM_ 64
#define NTOK_   (B_ * S_)   // 65536

typedef __attribute__((ext_vector_type(16))) __bf16 v16bf;
typedef __attribute__((ext_vector_type(8)))  float  v8f;

__device__ __forceinline__ __bf16 f2bf(float f) { return (__bf16)f; }

__device__ __forceinline__ v8f wmma_bf16(v16bf a, v16bf b, v8f c) {
  // (neg_a, A, neg_b, B, c_mod, C, reuse_a, reuse_b)
  return __builtin_amdgcn_wmma_f32_16x16x32_bf16(false, a, false, b, (short)0, c, false, false);
}

// Fragment gather matching the CDNA5 16-bit A/B layout:
// lanes 0-15 hold K {0..7, 16..23}; lanes 16-31 hold K {8..15, 24..31}.
// `row` points at a 32-element (padded to 40) K-row in LDS for this lane's
// M-row (A) or N-column (B; B column n == W row n since C = A @ W^T).
// Row stride 40 (80 B) keeps the 16 B chunks aligned and banks spread.
__device__ __forceinline__ v16bf load_frag(const __bf16* __restrict__ row, int lane) {
  const int kb = (lane & 16) ? 8 : 0;
  v16bf f;
#pragma unroll
  for (int i = 0; i < 8; ++i) f[i] = row[kb + i];
#pragma unroll
  for (int i = 0; i < 8; ++i) f[8 + i] = row[kb + 16 + i];
  return f;
}

// ---------------------------------------------------------------------------
// Core MMA loop: block tile 128(M) x 128(N), K-step 32.
// 8 waves as 4(M) x 2(N); wave tile 32x64: 2 A-frags x 4 B-frags = 8 WMMA.
// Software pipeline: regs hold tile k+1 while WMMAs consume LDS tile k.
// addrA/addrB: (row-in-tile, k) -> global pointer.
// ---------------------------------------------------------------------------
template <typename PA, typename PB>
__device__ __forceinline__ void mma_loop(int K, PA addrA, PB addrB,
                                         __bf16 (*As)[40], __bf16 (*Bs)[40],
                                         v8f acc[2][4], int tid)
{
  const int lane = tid & 31;
  const int wave = tid >> 5;
  const int mr = (wave >> 1) * 32;
  const int nr = (wave & 1) * 64;
  const int lh = lane & 15;
  const int r = tid >> 1;          // 0..127 (tile row)
  const int c = (tid & 1) * 16;    // 0 or 16 (k offset)

  float ra[16], rb[16];
#pragma unroll
  for (int i = 0; i < 16; ++i) {
    ra[i] = *addrA(r, c + i);
    rb[i] = *addrB(r, c + i);
  }

  for (int k0 = 0; k0 < K; k0 += 32) {
    __syncthreads();               // previous tile's consumers are done
#pragma unroll
    for (int i = 0; i < 16; ++i) {
      As[r][c + i] = f2bf(ra[i]);
      Bs[r][c + i] = f2bf(rb[i]);
    }
    if (k0 + 32 < K) {             // issue next tile's loads before compute
#pragma unroll
      for (int i = 0; i < 16; ++i) {
        ra[i] = *addrA(r, k0 + 32 + c + i);
        rb[i] = *addrB(r, k0 + 32 + c + i);
      }
    }
    if (k0 + 64 < K) {             // hint k+2 tile into cache
      __builtin_prefetch(addrA(r, k0 + 64 + c), 0, 3);
      __builtin_prefetch(addrB(r, k0 + 64 + c), 0, 3);
    }
    __syncthreads();               // staged tile visible
    v16bf fa[2], fb[4];
#pragma unroll
    for (int i = 0; i < 2; ++i) fa[i] = load_frag(&As[mr + 16 * i + lh][0], lane);
#pragma unroll
    for (int j = 0; j < 4; ++j) fb[j] = load_frag(&Bs[nr + 16 * j + lh][0], lane);
#pragma unroll
    for (int i = 0; i < 2; ++i)
#pragma unroll
      for (int j = 0; j < 4; ++j)
        acc[i][j] = wmma_bf16(fa[i], fb[j], acc[i][j]);
  }
}

// C/D layout: acc[e] holds (M = e + 8*(lane>=16), N = lane&15)
__device__ __forceinline__ void store_tile(float* __restrict__ C, int ldc,
                                           const float* __restrict__ bias,
                                           v8f acc[2][4], int mbase, int nbase,
                                           int tid)
{
  const int lane = tid & 31;
  const int wave = tid >> 5;
  const int mr = (wave >> 1) * 32;
  const int nr = (wave & 1) * 64;
  const int lh = lane & 15;
  const int rowOff = (lane >> 4) << 3;
#pragma unroll
  for (int i = 0; i < 2; ++i)
#pragma unroll
    for (int j = 0; j < 4; ++j) {
      const int col = nbase + nr + 16 * j + lh;
      const float b = bias[col];
#pragma unroll
      for (int e = 0; e < 8; ++e) {
        const int row = mbase + mr + 16 * i + rowOff + e;
        C[(size_t)row * ldc + col] = acc[i][j][e] + b;
      }
    }
}

// ---------------------------------------------------------------------------
// Generic C[M,N] = A[M,K] @ W[N,K]^T + bias. Grid (N/128, M/128), block 256.
// ---------------------------------------------------------------------------
__global__ __launch_bounds__(256)
void gemm_bias_kernel(const float* __restrict__ A, int lda,
                      const float* __restrict__ W, int ldw,
                      const float* __restrict__ bias,
                      float* __restrict__ C, int ldc, int K)
{
  __shared__ __align__(16) __bf16 As[128][40];
  __shared__ __align__(16) __bf16 Bs[128][40];

  const int tid = threadIdx.x;
  const int nbase = blockIdx.x * 128;
  const int mbase = blockIdx.y * 128;

  v8f acc[2][4] = {};
  mma_loop(K,
           [&](int r, int k) { return &A[(size_t)(mbase + r) * lda + k]; },
           [&](int r, int k) { return &W[(size_t)(nbase + r) * ldw + k]; },
           As, Bs, acc, tid);
  store_tile(C, ldc, bias, acc, mbase, nbase, tid);
}

// ---------------------------------------------------------------------------
// Fused: gate = sigmoid([x,query] @ gate_w^T + gate_b)        (K = 1280)
//        retr = r1 @ op_w^T + op_b                            (K = 256)
//        out  = gate*x + (1-gate)*retr ; gate-mean reduction.
// Grid: (1024/128, NTOK/128), block 256.
// ---------------------------------------------------------------------------
__global__ __launch_bounds__(256)
void gate_retr_fused_kernel(const float* __restrict__ x,
                            const float* __restrict__ query,
                            const float* __restrict__ r1,
                            const float* __restrict__ gate_w,
                            const float* __restrict__ gate_b,
                            const float* __restrict__ op_w,
                            const float* __restrict__ op_b,
                            float* __restrict__ out,
                            float* __restrict__ gate_sum)
{
  __shared__ __align__(16) __bf16 As[128][40];
  __shared__ __align__(16) __bf16 Bs[128][40];
  __shared__ float sGate;

  const int tid = threadIdx.x;
  if (tid == 0) sGate = 0.f;
  const int lane = tid & 31;
  const int wave = tid >> 5;
  const int mr = (wave >> 1) * 32;
  const int nr = (wave & 1) * 64;
  const int lh = lane & 15;
  const int nbase = blockIdx.x * 128;
  const int mbase = blockIdx.y * 128;

  // Phase 1: gate logits, K = 1280, A = concat(x, query) synthesized in loader
  v8f accG[2][4] = {};
  mma_loop(D_IN_ + M_SIZE_,
           [&](int r, int k) {
             return (k < D_IN_)
                 ? &x[(size_t)(mbase + r) * D_IN_ + k]
                 : &query[(size_t)(mbase + r) * M_SIZE_ + (k - D_IN_)];
           },
           [&](int r, int k) {
             return &gate_w[(size_t)(nbase + r) * (D_IN_ + M_SIZE_) + k];
           },
           As, Bs, accG, tid);

  // Phase 2: retrieved = r1 @ op_w^T, K = 256 (reuse LDS)
  v8f accR[2][4] = {};
  mma_loop(M_SIZE_,
           [&](int r, int k) { return &r1[(size_t)(mbase + r) * M_SIZE_ + k]; },
           [&](int r, int k) { return &op_w[(size_t)(nbase + r) * M_SIZE_ + k]; },
           As, Bs, accR, tid);

  // Epilogue: sigmoid, blend with x, write out, gate-mean partial
  const int rowOff = (lane >> 4) << 3;
  float gacc = 0.f;
#pragma unroll
  for (int i = 0; i < 2; ++i)
#pragma unroll
    for (int j = 0; j < 4; ++j) {
      const int col = nbase + nr + 16 * j + lh;
      const float gb = gate_b[col];
      const float ob = op_b[col];
#pragma unroll
      for (int e = 0; e < 8; ++e) {
        const int row = mbase + mr + 16 * i + rowOff + e;
        const float g  = 1.f / (1.f + __expf(-(accG[i][j][e] + gb)));
        const float rv = accR[i][j][e] + ob;
        const float xv = x[(size_t)row * D_IN_ + col];
        out[(size_t)row * D_IN_ + col] = g * xv + (1.f - g) * rv;
        gacc += g;
      }
    }
#pragma unroll
  for (int o = 16; o; o >>= 1) gacc += __shfl_xor(gacc, o, 32);
  if (lane == 0) atomicAdd(&sGate, gacc);
  __syncthreads();
  if (tid == 0)
    atomicAdd(gate_sum, sGate * (1.f / ((float)NTOK_ * (float)D_IN_)));
}

// ---------------------------------------------------------------------------
// Memory-bank projections (tiny: 16 x 256)
// ---------------------------------------------------------------------------
__global__ __launch_bounds__(256)
void kv_proj_kernel(const float* __restrict__ mb, const float* __restrict__ strength,
                    const float* __restrict__ kp_w, const float* __restrict__ kp_b,
                    const float* __restrict__ vp_w, const float* __restrict__ vp_b,
                    float* __restrict__ keys, float* __restrict__ values)
{
  __shared__ float sMB[M_SIZE_];
  const int m = blockIdx.x, j = threadIdx.x;
  sMB[j] = mb[m * M_SIZE_ + j];
  __syncthreads();
  const float* kr = &kp_w[(size_t)j * M_SIZE_];
  const float* vr = &vp_w[(size_t)j * M_SIZE_];
  float sk = 0.f, sv = 0.f;
  for (int i = 0; i < M_SIZE_; ++i) { sk += sMB[i] * kr[i]; sv += sMB[i] * vr[i]; }
  keys[m * M_SIZE_ + j]   = sk + kp_b[j];
  values[m * M_SIZE_ + j] = (sv + vp_b[j]) * strength[m];
}

__global__ __launch_bounds__(256)
void kv_head_proj_kernel(const float* __restrict__ keys, const float* __restrict__ values,
                         const float* __restrict__ in_w, const float* __restrict__ in_b,
                         float* __restrict__ kk, float* __restrict__ vv)
{
  __shared__ float sKe[M_SIZE_];
  __shared__ float sVa[M_SIZE_];
  const int m = blockIdx.x, j = threadIdx.x;
  sKe[j] = keys[m * M_SIZE_ + j];
  sVa[j] = values[m * M_SIZE_ + j];
  __syncthreads();
  const float* wk = &in_w[(size_t)(M_SIZE_ + j) * M_SIZE_];
  const float* wv = &in_w[(size_t)(2 * M_SIZE_ + j) * M_SIZE_];
  float a = 0.f, b = 0.f;
  for (int i = 0; i < M_SIZE_; ++i) { a += sKe[i] * wk[i]; b += sVa[i] * wv[i]; }
  kk[m * M_SIZE_ + j] = a + in_b[M_SIZE_ + j];
  vv[m * M_SIZE_ + j] = b + in_b[2 * M_SIZE_ + j];
}

// ---------------------------------------------------------------------------
// Attention over 16 memory cells. One wave per token, 8 tokens per block.
// Also accumulates attn_weights.mean(axis=(heads, s)) -> [B, 16].
// ---------------------------------------------------------------------------
__global__ __launch_bounds__(256)
void attention_kernel(const float* __restrict__ q,
                      const float* __restrict__ kk,
                      const float* __restrict__ vv,
                      float* __restrict__ ctx,
                      float* __restrict__ attn_out)
{
  __shared__ float sK[M_CELLS_][M_SIZE_];
  __shared__ float sV[M_CELLS_][M_SIZE_];
  __shared__ float sQ[8][M_SIZE_];
  __shared__ float sAttn[8][64];
  __shared__ float sBA[M_CELLS_];

  const int tid = threadIdx.x;
  const int lane = tid & 31;
  const int wave = tid >> 5;
  const int t0 = blockIdx.x * 8;           // 8 tokens, all in same batch b

  if (tid < M_CELLS_) sBA[tid] = 0.f;
  for (int i = tid; i < M_CELLS_ * M_SIZE_; i += 256) {
    sK[i >> 8][i & 255] = kk[i];
    sV[i >> 8][i & 255] = vv[i];
  }
  for (int i = tid; i < 8 * M_SIZE_; i += 256)
    sQ[i >> 8][i & 255] = q[(size_t)t0 * M_SIZE_ + i];
  __syncthreads();

  // each lane owns score pairs p = lane (h=0,1) and p = lane+32 (h=2,3); p = h*16+m
#pragma unroll
  for (int s = 0; s < 2; ++s) {
    const int p = lane + 32 * s;
    const int h = p >> 4, m = p & 15;
    const float* qr = &sQ[wave][h * HEAD_DIM_];
    const float* kr = &sK[m][h * HEAD_DIM_];
    float sc = 0.f;
#pragma unroll 8
    for (int d = 0; d < HEAD_DIM_; ++d) sc += qr[d] * kr[d];
    sc *= 0.125f;                          // 1/sqrt(64)
    // softmax across the 16 lanes sharing h (wave32: width-16 groups)
    float mx = sc;
#pragma unroll
    for (int o = 8; o; o >>= 1) mx = fmaxf(mx, __shfl_xor(mx, o, 16));
    const float e = __expf(sc - mx);
    float sum = e;
#pragma unroll
    for (int o = 8; o; o >>= 1) sum += __shfl_xor(sum, o, 16);
    const float a = e / sum;
    sAttn[wave][p] = a;
    atomicAdd(&sBA[m], a);
  }
  __syncthreads();

  // ctx[j] = sum_m attn[h(j), m] * v[m][j]
#pragma unroll
  for (int cchunk = 0; cchunk < 8; ++cchunk) {
    const int j = cchunk * 32 + lane;
    const int h = j >> 6;
    float acc = 0.f;
#pragma unroll
    for (int m = 0; m < M_CELLS_; ++m) acc += sAttn[wave][h * 16 + m] * sV[m][j];
    ctx[(size_t)(t0 + wave) * M_SIZE_ + j] = acc;
  }

  if (tid < M_CELLS_) {
    const int b = t0 >> 11;                // / S_
    atomicAdd(&attn_out[b * M_CELLS_ + tid],
              sBA[tid] * (1.f / ((float)N_HEADS_ * (float)S_)));
  }
}

// ---------------------------------------------------------------------------
// In-place LayerNorm over D_IN, one block per row.
// ---------------------------------------------------------------------------
__global__ __launch_bounds__(256)
void layernorm_kernel(float* __restrict__ out,
                      const float* __restrict__ g, const float* __restrict__ b)
{
  __shared__ float red[2][8];
  const int tid = threadIdx.x;
  float* rp = out + (size_t)blockIdx.x * D_IN_;
  float v[4];
  float s = 0.f, s2 = 0.f;
#pragma unroll
  for (int i = 0; i < 4; ++i) {
    v[i] = rp[tid + i * 256];
    s += v[i]; s2 += v[i] * v[i];
  }
#pragma unroll
  for (int o = 16; o; o >>= 1) { s += __shfl_xor(s, o, 32); s2 += __shfl_xor(s2, o, 32); }
  const int lane = tid & 31, wave = tid >> 5;
  if (lane == 0) { red[0][wave] = s; red[1][wave] = s2; }
  __syncthreads();
  if (wave == 0) {
    s  = (lane < 8) ? red[0][lane] : 0.f;
    s2 = (lane < 8) ? red[1][lane] : 0.f;
#pragma unroll
    for (int o = 4; o; o >>= 1) { s += __shfl_xor(s, o, 8); s2 += __shfl_xor(s2, o, 8); }
    if (lane == 0) { red[0][0] = s; red[1][0] = s2; }
  }
  __syncthreads();
  const float mu  = red[0][0] * (1.f / D_IN_);
  const float var = red[1][0] * (1.f / D_IN_) - mu * mu;
  const float inv = rsqrtf(var + 1e-5f);
#pragma unroll
  for (int i = 0; i < 4; ++i) {
    const int j = tid + i * 256;
    rp[j] = (v[i] - mu) * inv * g[j] + b[j];
  }
}

__global__ void init_tail_kernel(float* __restrict__ tail) {
  if (threadIdx.x < B_ * M_CELLS_ + 1) tail[threadIdx.x] = 0.f;
}

// ---------------------------------------------------------------------------
extern "C" void kernel_launch(void* const* d_in, const int* in_sizes, int n_in,
                              void* d_out, int out_size, void* d_ws, size_t ws_size,
                              hipStream_t stream)
{
  const float* x    = (const float*)d_in[0];
  const float* mb   = (const float*)d_in[1];
  const float* mstr = (const float*)d_in[2];
  const float* qp_w = (const float*)d_in[3];
  const float* qp_b = (const float*)d_in[4];
  const float* kp_w = (const float*)d_in[5];
  const float* kp_b = (const float*)d_in[6];
  const float* vp_w = (const float*)d_in[7];
  const float* vp_b = (const float*)d_in[8];
  const float* in_w = (const float*)d_in[9];
  const float* in_b = (const float*)d_in[10];
  const float* ao_w = (const float*)d_in[11];
  const float* ao_b = (const float*)d_in[12];
  const float* op_w = (const float*)d_in[13];
  const float* op_b = (const float*)d_in[14];
  const float* g_w  = (const float*)d_in[15];
  const float* g_b  = (const float*)d_in[16];
  const float* lng  = (const float*)d_in[17];
  const float* lnb  = (const float*)d_in[18];

  float* out      = (float*)d_out;
  float* attn_out = out + (size_t)NTOK_ * D_IN_;        // [32,16]
  float* gate_out = attn_out + B_ * M_CELLS_;           // scalar

  // workspace layout (~192 MB + 64 KB)
  float* wsQuery = (float*)d_ws;                        // NTOK*256 (kept for gate)
  float* wsQ     = wsQuery + (size_t)NTOK_ * M_SIZE_;   // q; reused as r1
  float* wsCtx   = wsQ + (size_t)NTOK_ * M_SIZE_;       // NTOK*256
  float* wsKeys  = wsCtx + (size_t)NTOK_ * M_SIZE_;
  float* wsVals  = wsKeys + M_CELLS_ * M_SIZE_;
  float* wsKK    = wsVals + M_CELLS_ * M_SIZE_;
  float* wsVV    = wsKK + M_CELLS_ * M_SIZE_;

  const dim3 blk(256);

  init_tail_kernel<<<1, 544, 0, stream>>>(attn_out);
  kv_proj_kernel<<<M_CELLS_, blk, 0, stream>>>(mb, mstr, kp_w, kp_b, vp_w, vp_b,
                                               wsKeys, wsVals);
  kv_head_proj_kernel<<<M_CELLS_, blk, 0, stream>>>(wsKeys, wsVals, in_w, in_b,
                                                    wsKK, wsVV);
  // query = x @ qp_w^T + qp_b   [NTOK,1024] x [256,1024]^T
  gemm_bias_kernel<<<dim3(M_SIZE_ / 128, NTOK_ / 128), blk, 0, stream>>>(
      x, D_IN_, qp_w, D_IN_, qp_b, wsQuery, M_SIZE_, D_IN_);
  // q = query @ wq^T + bq       (wq = first 256 rows of in_w)
  gemm_bias_kernel<<<dim3(M_SIZE_ / 128, NTOK_ / 128), blk, 0, stream>>>(
      wsQuery, M_SIZE_, in_w, M_SIZE_, in_b, wsQ, M_SIZE_, M_SIZE_);
  attention_kernel<<<NTOK_ / 8, blk, 0, stream>>>(wsQ, wsKK, wsVV, wsCtx, attn_out);
  // r1 = ctx @ attn_out_w^T + attn_out_b   (overwrites dead q buffer)
  gemm_bias_kernel<<<dim3(M_SIZE_ / 128, NTOK_ / 128), blk, 0, stream>>>(
      wsCtx, M_SIZE_, ao_w, M_SIZE_, ao_b, wsQ, M_SIZE_, M_SIZE_);
  // fused gate GEMM (K=1280) + op GEMM (K=256) + sigmoid blend + gate mean
  gate_retr_fused_kernel<<<dim3(D_IN_ / 128, NTOK_ / 128), blk, 0, stream>>>(
      x, wsQuery, wsQ, g_w, g_b, op_w, op_b, out, gate_out);
  layernorm_kernel<<<NTOK_, blk, 0, stream>>>(out, lng, lnb);
}
